// HierarchicalRefinementQuantizer_82617990906585
// MI455X (gfx1250) — matrix operand order repaired and checked by reading, not compiled, not run on hardware
//
#include <hip/hip_runtime.h>
#include <hip/hip_bf16.h>

// Problem constants (from reference): B=8192, D=1024, K=4096, H=8
#define BQ   8192
#define DQ   1024
#define KQ   4096
#define HQ   8
#define BM   16      // rows per block (shared by all waves)
#define NT   (KQ/16) // 256 n-tiles of 16 codebook entries

typedef __bf16 bf16_t;
typedef __bf16 v16bf __attribute__((ext_vector_type(16)));
typedef __bf16 v8bf  __attribute__((ext_vector_type(8)));
typedef float  v8f   __attribute__((ext_vector_type(8)));

// ---- bf16 helpers (bit-level RNE, no scalar __bf16 arithmetic needed) ----
__device__ __forceinline__ unsigned short f2bf_rne(float f) {
  unsigned u = __float_as_uint(f);
  unsigned r = u + 0x7fffu + ((u >> 16) & 1u);
  return (unsigned short)(r >> 16);
}
__device__ __forceinline__ float bf2f(unsigned short b) {
  return __uint_as_float(((unsigned)b) << 16);
}

// =====================================================================
// Kernel 1: transpose codebooks [H][K][D] f32 -> Et_hi/Et_lo [H][D][K] bf16
// (k contiguous so WMMA B-fragment loads are coalesced), split hi/lo.
// grid = (K/32, D/32, H), block = (32, 8)
// =====================================================================
__global__ void hrq_prep_transpose_split(const float* __restrict__ cb,
                                         unsigned short* __restrict__ et_hi,
                                         unsigned short* __restrict__ et_lo) {
  __shared__ float tile[32][33];
  const int h  = blockIdx.z;
  const int kt = blockIdx.x * 32;
  const int dt = blockIdx.y * 32;
  const int tx = threadIdx.x, ty = threadIdx.y;
  const float* src = cb + ((size_t)h * KQ) * DQ;
#pragma unroll
  for (int p = 0; p < 4; ++p)
    tile[ty + 8 * p][tx] = src[(size_t)(kt + ty + 8 * p) * DQ + dt + tx];
  __syncthreads();
#pragma unroll
  for (int p = 0; p < 4; ++p) {
    const int dl = ty + 8 * p;           // local d
    const float v = tile[tx][dl];        // = E[kt+tx][dt+dl]
    const unsigned short hi = f2bf_rne(v);
    const unsigned short lo = f2bf_rne(v - bf2f(hi));
    const size_t o = ((size_t)h * DQ + dt + dl) * KQ + kt + tx;
    et_hi[o] = hi;
    et_lo[o] = lo;
  }
}

// =====================================================================
// Kernel 2: halfnorm[h*K+k] = 0.5*sum_d E^2 ; one wave32 per row.
// grid = H*K/8, block = 256
// =====================================================================
__global__ void hrq_prep_halfnorms(const float* __restrict__ cb,
                                   float* __restrict__ hn) {
  const int row  = blockIdx.x * 8 + (threadIdx.x >> 5);
  const int lane = threadIdx.x & 31;
  const float* r = cb + (size_t)row * DQ;
  float s = 0.f;
  for (int d = lane; d < DQ; d += 32) { float v = r[d]; s += v * v; }
#pragma unroll
  for (int m = 16; m >= 1; m >>= 1) s += __shfl_xor(s, m, 32);
  if (lane == 0) hn[row] = 0.5f * s;
}

// =====================================================================
// Kernel 3: one head of the cascade.
//  block: 256 threads = 8 waves; 16 residual rows per block.
//  Phase A: stage rows to LDS as bf16 hi/lo planes (once).
//  Phase B: each wave owns a 16x64 output strip per outer step (4 n-tiles,
//           4 accumulators) -> every A fragment fetched from LDS feeds 12
//           v_wmma_f32_16x16x32_bf16 (split-bf16: hi*hi + lo*hi + hi*lo),
//           4x less LDS traffic per WMMA than 1-tile blocking.
//  Phase C: cross-wave argmax reduce (first-index ties), exact-f32 gather,
//           residual + quantized update, code write.
// =====================================================================
__global__ __launch_bounds__(256)
void hrq_head_pass(const float* __restrict__ src,        // resid source (inputs if h==0)
                   float* __restrict__ resid,            // resid out (ws)
                   const float* __restrict__ cb_h,       // exact f32 codebook, this head
                   const unsigned short* __restrict__ eth, // Et_hi, this head [D][K]
                   const unsigned short* __restrict__ etl, // Et_lo, this head [D][K]
                   const float* __restrict__ hn,         // halfnorms, this head [K]
                   float* __restrict__ outq,             // quantized [B][D]
                   float* __restrict__ outc,             // codes [B][H] (as float)
                   int h) {
  __shared__ unsigned a_hi[BM * (DQ / 2)];  // 16 rows x 1024 bf16 (packed u32) = 32KB
  __shared__ unsigned a_lo[BM * (DQ / 2)];  // 32KB
  __shared__ float red_val[8][BM][16];
  __shared__ int   red_idx[8][BM][16];
  __shared__ int   s_best[BM];

  const int tid  = threadIdx.x;
  const int wave = tid >> 5;
  const int lane = tid & 31;
  const size_t row0 = (size_t)blockIdx.x * BM;

  // ---- Phase A: stage residual rows to LDS, split into bf16 hi/lo ----
  for (int i = tid; i < BM * (DQ / 2); i += 256) {
    const int r = i >> 9;         // row (DQ/2 = 512 u32 per row)
    const int c = i & 511;        // u32 column (2 d's)
    const float f0 = src[(row0 + r) * DQ + 2 * c + 0];
    const float f1 = src[(row0 + r) * DQ + 2 * c + 1];
    const unsigned short h0 = f2bf_rne(f0), h1 = f2bf_rne(f1);
    const unsigned short l0 = f2bf_rne(f0 - bf2f(h0));
    const unsigned short l1 = f2bf_rne(f1 - bf2f(h1));
    a_hi[i] = (unsigned)h0 | ((unsigned)h1 << 16);
    a_lo[i] = (unsigned)l0 | ((unsigned)l1 << 16);
  }
  __syncthreads();

  // A-fragment addressing (16-bit A 16x32): lane m<16 -> row m, K 0..7/16..23;
  // lanes 16..31 -> row m, K 8..15/24..31.
  const int am  = lane & 15;
  const int sel = lane >> 4;
  const unsigned* arow_hi = a_hi + am * (DQ / 2);
  const unsigned* arow_lo = a_lo + am * (DQ / 2);

  float best_v[8];
  int   best_k[8];
#pragma unroll
  for (int i = 0; i < 8; ++i) { best_v[i] = -3.402823466e38f; best_k[i] = 0; }

  // ---- Phase B: 8 outer steps, each wave computes a 16x64 strip ----
  for (int it = 0; it < NT / (8 * 4); ++it) {       // 8 steps
    const int n0 = (it * 8 + wave) * 64;            // 64 columns per strip
    // B: lane -> column n = lane&15; elements over d = kc*32 + 8*sel + {0..7,16..23}
    // (row stride K), matching the A-fragment k order.
    const unsigned short* bph = eth + (size_t)(sel * 8) * KQ + n0 + (lane & 15);
    const unsigned short* bpl = etl + (size_t)(sel * 8) * KQ + n0 + (lane & 15);
    __builtin_prefetch(bph + 8 * 64, 0, 1);         // next step's strip (L2 hint)
    v8f acc[4] = {};
#pragma unroll 2
    for (int kc = 0; kc < 32; ++kc) {
      v16bf afh, afl;
      const int o1 = kc * 16 + sel * 4;   // u32 offset of first 8-bf16 chunk
      ((v8bf*)&afh)[0] = *(const v8bf*)(arow_hi + o1);
      ((v8bf*)&afh)[1] = *(const v8bf*)(arow_hi + o1 + 8);
      ((v8bf*)&afl)[0] = *(const v8bf*)(arow_lo + o1);
      ((v8bf*)&afl)[1] = *(const v8bf*)(arow_lo + o1 + 8);
      const unsigned short* bh = bph + (size_t)kc * 32 * KQ;
      const unsigned short* bl = bpl + (size_t)kc * 32 * KQ;
#pragma unroll
      for (int t = 0; t < 4; ++t) {
        v16bf bfh, bfl;
#pragma unroll
        for (int s = 0; s < 16; ++s) {
          const size_t ko = (size_t)(s < 8 ? s : s + 8) * KQ + t * 16;
          bfh[s] = *(const bf16_t*)(bh + ko);
          bfl[s] = *(const bf16_t*)(bl + ko);
        }
        // split-bf16 fp32-accurate product: hi*hi + lo*hi + hi*lo
        acc[t] = __builtin_amdgcn_wmma_f32_16x16x32_bf16(false, afh, false, bfh,
                                                         (short)0, acc[t], false, false);
        acc[t] = __builtin_amdgcn_wmma_f32_16x16x32_bf16(false, afl, false, bfh,
                                                         (short)0, acc[t], false, false);
        acc[t] = __builtin_amdgcn_wmma_f32_16x16x32_bf16(false, afh, false, bfl,
                                                         (short)0, acc[t], false, false);
      }
    }
    // C layout: lane<16 -> N=lane,  M=i ; lane>=16 -> N=lane-16, M=i+8
#pragma unroll
    for (int t = 0; t < 4; ++t) {
      const int ncol = n0 + t * 16 + (lane & 15);
      const float b  = hn[ncol];
#pragma unroll
      for (int i = 0; i < 8; ++i) {
        const float v = acc[t][i] - b;  // argmax(r.E - 0.5||E||^2) == argmin dist
        if (v > best_v[i]) { best_v[i] = v; best_k[i] = ncol; }  // ncol strictly
      }                                                           // increasing ->
    }                                                             // first-idx ties
  }

  // ---- Phase C: cross-lane / cross-wave argmax reduce ----
#pragma unroll
  for (int i = 0; i < 8; ++i) {
    const int m = (lane < 16) ? i : (i + 8);
    red_val[wave][m][lane & 15] = best_v[i];
    red_idx[wave][m][lane & 15] = best_k[i];
  }
  __syncthreads();
  if (tid < BM) {
    float bv = -3.402823466e38f;
    int   bk = 0x7fffffff;
    for (int w = 0; w < 8; ++w)
      for (int s = 0; s < 16; ++s) {
        const float v = red_val[w][tid][s];
        const int   k = red_idx[w][tid][s];
        if (v > bv || (v == bv && k < bk)) { bv = v; bk = k; }
      }
    s_best[tid] = bk;
    outc[(row0 + tid) * HQ + h] = (float)bk;   // code for this head
  }
  __syncthreads();

  // exact-f32 gather + residual/quantized update (block owns its rows)
  for (int i = tid; i < BM * DQ; i += 256) {
    const int r = i >> 10;
    const int d = i & (DQ - 1);
    const int bk = s_best[r];
    const float q  = cb_h[(size_t)bk * DQ + d];
    const size_t o = (row0 + r) * DQ + d;
    resid[o] = src[o] - q;
    if (h == 0) outq[o] = q; else outq[o] += q;
  }
}

// =====================================================================
// Host launcher.  ws layout (needs ~161 MB):
//   [0, 64MB)      Et_hi bf16 [H][D][K]
//   [64, 128MB)    Et_lo bf16 [H][D][K]
//   [+128KB)       halfnorms f32 [H][K]
//   [+32MB)        resid f32 [B][D]
// =====================================================================
extern "C" void kernel_launch(void* const* d_in, const int* in_sizes, int n_in,
                              void* d_out, int out_size, void* d_ws, size_t ws_size,
                              hipStream_t stream) {
  const float* inputs    = (const float*)d_in[0];   // [B][1][D]
  const float* codebooks = (const float*)d_in[1];   // [H][K][D]
  float* outq = (float*)d_out;                      // quantized [B][D]
  float* outc = outq + (size_t)BQ * DQ;             // codes [B][H]

  char* ws = (char*)d_ws;
  const size_t ETN = (size_t)HQ * DQ * KQ;          // 33,554,432 elems
  unsigned short* et_hi = (unsigned short*)ws;
  unsigned short* et_lo = (unsigned short*)(ws + ETN * 2);
  float* hnorm = (float*)(ws + ETN * 4);
  float* resid = (float*)(ws + ETN * 4 + (size_t)HQ * KQ * 4);

  // 1) transpose + hi/lo split of codebooks
  hrq_prep_transpose_split<<<dim3(KQ / 32, DQ / 32, HQ), dim3(32, 8), 0, stream>>>(
      codebooks, et_hi, et_lo);
  // 2) half-norms
  hrq_prep_halfnorms<<<(HQ * KQ) / 8, 256, 0, stream>>>(codebooks, hnorm);
  // 3) sequential cascade over heads
  for (int h = 0; h < HQ; ++h) {
    const float* src = (h == 0) ? inputs : resid;
    hrq_head_pass<<<BQ / BM, 256, 0, stream>>>(
        src, resid,
        codebooks + (size_t)h * KQ * DQ,
        et_hi + (size_t)h * DQ * KQ,
        et_lo + (size_t)h * DQ * KQ,
        hnorm + (size_t)h * KQ,
        outq, outc, h);
  }
}
